// MpMaxPoolingMatch_53584011985091
// MI455X (gfx1250) — matrix operand order repaired and checked by reading, not compiled, not run on hardware
//
#include <hip/hip_runtime.h>

typedef __attribute__((ext_vector_type(16))) _Float16 v16h;
typedef __attribute__((ext_vector_type(8)))  _Float16 v8h;
typedef __attribute__((ext_vector_type(4)))  _Float16 v4h;
typedef __attribute__((ext_vector_type(8)))  float    v8f;
typedef __attribute__((ext_vector_type(4)))  float    v4f;

#define B_   16
#define T1_  256
#define T2_  256
#define D_   256
#define MP_  32

#define LTS  264   // LDS row stride in halfs: 528 bytes, 16B-aligned, rows offset 4 banks

// B fragment k order chosen here (32x16, same as A): lane half h holds 16 contiguous
// K values (K = base + 16h + e). Caller passes p = row + base + 16*h (f32 src).
__device__ __forceinline__ v16h frag_b_f32(const float* p) {
  v4f x0 = *(const v4f*)(p);
  v4f x1 = *(const v4f*)(p + 4);
  v4f x2 = *(const v4f*)(p + 8);
  v4f x3 = *(const v4f*)(p + 12);
  v16h f;
#pragma unroll
  for (int t = 0; t < 4; ++t) {
    f[t]      = (_Float16)x0[t];
    f[4 + t]  = (_Float16)x1[t];
    f[8 + t]  = (_Float16)x2[t];
    f[12 + t] = (_Float16)x3[t];
  }
  return f;
}

// Cooperative stage of one 16x256 f32 lt tile into LDS as f16 (converted once).
__device__ __forceinline__ void stage_tile(const float* __restrict__ src,
                                           _Float16 (*dst)[LTS], int t) {
#pragma unroll
  for (int q = 0; q < 4; ++q) {
    int idx = t + q * 256;       // float4 index, 0..1023
    int row = idx >> 6;          // 64 float4 per 256-float row
    int c4  = idx & 63;
    v4f x = *(const v4f*)(src + row * D_ + c4 * 4);
    v4h hv;
    hv[0] = (_Float16)x[0]; hv[1] = (_Float16)x[1];
    hv[2] = (_Float16)x[2]; hv[3] = (_Float16)x[3];
    *(v4h*)(&dst[row][c4 * 4]) = hv;   // ds_store_b64, 8B aligned
  }
}

__global__ __launch_bounds__(256) void MpMaxPoolingMatch_wmma_kernel(
    const float* __restrict__ lt,   // [B, T1, D]
    const float* __restrict__ rt,   // [B, T2, D]
    const float* __restrict__ kw,   // [MP, D]
    float* __restrict__ out)        // [B, T2, MP]
{
  __shared__ _Float16 slt[2][16][LTS];   // double-buffered f16 lt tile (~17 KB)

  const int t    = threadIdx.x;
  const int lane = t & 31;
  const int wave = t >> 5;
  const int r    = lane & 15;
  const int h    = lane >> 4;
  const int j0   = blockIdx.x * 16;              // rt column tile
  const int m0   = blockIdx.y * 16 + wave * 2;   // two kernel rows per wave
  const int m1   = m0 + 1;
  const int b    = blockIdx.z;

  // ---- Setup: B' = (rt ⊙ K[m]) fragments, loop-invariant over i ----
  const float* rtrow = rt + (size_t)(b * T2_ + j0 + r) * D_ + 16 * h;
  const float* k0row = kw + (size_t)m0 * D_ + 16 * h;
  const float* k1row = k0row + D_;
  v16h bf0[8], bf1[8];
#pragma unroll
  for (int kb = 0; kb < 8; ++kb) {
    v16h bv = frag_b_f32(rtrow + kb * 32);
    bf0[kb] = bv * frag_b_f32(k0row + kb * 32);   // v_pk_mul_f16, once
    bf1[kb] = bv * frag_b_f32(k1row + kb * 32);
  }

  float acc0[8], acc1[8];
#pragma unroll
  for (int v = 0; v < 8; ++v) { acc0[v] = -3.402823466e38f; acc1[v] = -3.402823466e38f; }

  const float* ltbase = lt + (size_t)b * T1_ * D_;

  // prologue: stage tile 0
  stage_tile(ltbase, slt[0], t);
  __syncthreads();

  for (int it = 0; it < 16; ++it) {
    const int cur = it & 1;

    // stage next tile into the other buffer while computing this one
    if (it + 1 < 16) stage_tile(ltbase + (it + 1) * 16 * D_, slt[cur ^ 1], t);
    if (it + 2 < 16) __builtin_prefetch(ltbase + (it + 2) * 16 * D_ + t * 16, 0, 3);

    // ---- 8 K-steps, 2 WMMAs each (shared A fragment, 2 accumulators) ----
    v8f c0 = {}, c1 = {};
    const _Float16* lrow = &slt[cur][r][16 * h];
#pragma unroll
    for (int kb = 0; kb < 8; ++kb) {
      // A layout: halves 0..15 at K=base+16h+e (contiguous, matches B)
      v8h lo = *(const v8h*)(lrow + kb * 32);        // ds_load_b128
      v8h hi = *(const v8h*)(lrow + kb * 32 + 8);    // ds_load_b128
      v16h a = __builtin_shufflevector(lo, hi, 0, 1, 2, 3, 4, 5, 6, 7,
                                               8, 9, 10, 11, 12, 13, 14, 15);
      c0 = __builtin_amdgcn_wmma_f32_16x16x32_f16(false, a, false, bf0[kb],
                                                  (short)0, c0, false, false);
      c1 = __builtin_amdgcn_wmma_f32_16x16x32_f16(false, a, false, bf1[kb],
                                                  (short)0, c1, false, false);
    }

    // fold this i-tile into the running max (tanh deferred: monotonic)
#pragma unroll
    for (int v = 0; v < 8; ++v) {
      acc0[v] = fmaxf(acc0[v], c0[v]);
      acc1[v] = fmaxf(acc1[v], c1[v]);
    }
    __syncthreads();
  }

  // ---- reduce max over 16 rows: 8 in-lane + lane-half combine ----
  float r0 = acc0[0], r1 = acc1[0];
#pragma unroll
  for (int v = 1; v < 8; ++v) { r0 = fmaxf(r0, acc0[v]); r1 = fmaxf(r1, acc1[v]); }
  r0 = fmaxf(r0, __shfl_xor(r0, 16, 32));
  r1 = fmaxf(r1, __shfl_xor(r1, 16, 32));

  if (lane < 16) {
    size_t o = (size_t)(b * T2_ + j0 + lane) * MP_;
    out[o + m0] = tanhf(r0);
    out[o + m1] = tanhf(r1);
  }
}

extern "C" void kernel_launch(void* const* d_in, const int* in_sizes, int n_in,
                              void* d_out, int out_size, void* d_ws, size_t ws_size,
                              hipStream_t stream) {
  const float* lt = (const float*)d_in[0];   // [16,256,256]
  const float* rt = (const float*)d_in[1];   // [16,256,256]
  const float* kw = (const float*)d_in[2];   // [32,256]
  float* out = (float*)d_out;                // [16,256,32]

  dim3 grid(T2_ / 16, MP_ / 16, B_);         // (16, 2, 16)
  dim3 block(256);                           // 8 waves, 2 m-values each
  MpMaxPoolingMatch_wmma_kernel<<<grid, block, 0, stream>>>(lt, rt, kw, out);
}
